// LinearAttention_10754598109284
// MI455X (gfx1250) — matrix operand order repaired and checked by reading, not compiled, not run on hardware
//
#include <hip/hip_runtime.h>
#include <hip/hip_bf16.h>

// ---------------------------------------------------------------------------
// LinearAttention for MI455X (gfx1250, wave32, WMMA).
// B=8, L=32768, CH=128, H=4, D=32.  All GEMMs on v_wmma_f32_16x16x32_f16.
// Reductions (GroupNorm stats) accumulate in f64; softmaxes in f32.
// ---------------------------------------------------------------------------

typedef __attribute__((ext_vector_type(16))) _Float16 v16h;
typedef __attribute__((ext_vector_type(8)))  _Float16 v8h;
typedef __attribute__((ext_vector_type(8)))  float    v8f;

constexpr int kB  = 8;
constexpr int kL  = 32768;
constexpr int kCH = 128;
constexpr int kH  = 4;
constexpr int kD  = 32;
constexpr int kQKV = 3 * kCH;            // 384
constexpr float kEPS    = 1e-5f;
constexpr float kKSCALE = 0.005524271728019903f;   // 1/sqrt(L)
constexpr float kQSCALE = 0.17677669529663687f;    // 1/sqrt(D)

// ---- WMMA helper -----------------------------------------------------------
__device__ inline v8f wmma_f16(v16h a, v16h b, v8f c) {
  // D = A(16x32 f16) * B(32x16 f16) + C(16x16 f32)
  return __builtin_amdgcn_wmma_f32_16x16x32_f16(
      /*neg_a=*/false, a, /*neg_b=*/false, b,
      /*c_mod=*/(short)0, c, /*reuse_a=*/false, /*reuse_b=*/false);
}

// A fragment (16x32, f16) from a row-major [16 x ld] source.
// Lane t: row M = t&15; K chunk offset = (t<16)?0:16 ; elements 0..15 -> K=koff..koff+15
// (contiguous order, matching the B-fragment packing).
__device__ inline v16h load_a_f16(const _Float16* base, int ld, int k0, int lane) {
  int m    = lane & 15;
  int koff = (lane < 16) ? 0 : 16;
  const _Float16* p = base + m * ld + k0 + koff;
  v8h lo = *(const v8h*)(p);
  v8h hi = *(const v8h*)(p + 8);
  v16h a;
#pragma unroll
  for (int i = 0; i < 8; ++i) { a[i] = lo[i]; a[8 + i] = hi[i]; }
  return a;
}

// Ordered-uint key for float atomic max
__device__ inline unsigned fkey(float f) {
  unsigned u = __float_as_uint(f);
  return (u & 0x80000000u) ? ~u : (u | 0x80000000u);
}
__device__ inline float funkey(unsigned u) {
  return __uint_as_float((u & 0x80000000u) ? (u & 0x7fffffffu) : ~u);
}

// ---------------------------------------------------------------------------
// K0: zero accumulators + pack weights into WMMA B-fragment order (f16).
// B-fragment pack index: ((kc*NT + nt)*512 + lane*16 + j) holds W[kc*32 + koff + j][nt*16 + n]
// with koff=(lane<16)?0:16, n=lane&15  (contiguous K order, same as load_a_f16).
// ---------------------------------------------------------------------------
__global__ __launch_bounds__(256) void init_kernel(
    const float* __restrict__ Wqkv, const float* __restrict__ Wout,
    _Float16* __restrict__ wqkvp, _Float16* __restrict__ woutp,
    float* __restrict__ ctx, float* __restrict__ sumP,
    unsigned* __restrict__ kmaxk, double* __restrict__ gnd) {
  int id = blockIdx.x * 256 + threadIdx.x;
  if (id < 4 * 24 * 512) {  // W_qkv: 128x384
    int kc = id / (24 * 512), r = id % (24 * 512);
    int nt = r / 512, r2 = r % 512, lane = r2 >> 4, j = r2 & 15;
    int k = kc * 32 + ((lane < 16) ? 0 : 16) + j;
    int n = nt * 16 + (lane & 15);
    wqkvp[id] = (_Float16)Wqkv[k * kQKV + n];
  }
  if (id < 4 * 8 * 512) {   // W_out: 128x128
    int kc = id / (8 * 512), r = id % (8 * 512);
    int nt = r / 512, r2 = r % 512, lane = r2 >> 4, j = r2 & 15;
    int k = kc * 32 + ((lane < 16) ? 0 : 16) + j;
    int n = nt * 16 + (lane & 15);
    woutp[id] = (_Float16)Wout[k * kCH + n];
  }
  if (id < kB * kH * kD * kD) ctx[id] = 0.f;
  if (id < kB * kH * kD) { sumP[id] = 0.f; kmaxk[id] = 0u; }
  if (id < 4 * kB) gnd[id] = 0.0;
}

// ---------------------------------------------------------------------------
// K1: GroupNorm1 stats (f64 accumulation, atomic f64 add per segment).
// ---------------------------------------------------------------------------
__global__ __launch_bounds__(256) void gn_stats_kernel(
    const float* __restrict__ x, double* __restrict__ gnd) {
  int b = blockIdx.y;
  const float* p = x + (size_t)b * kL * kCH + (size_t)blockIdx.x * 65536;
  double s = 0.0, ss = 0.0;
  for (int i = threadIdx.x; i < 65536; i += 256) {
    float v = p[i];
    s += (double)v; ss += (double)v * (double)v;
  }
  __shared__ double shS[256], shQ[256];
  shS[threadIdx.x] = s; shQ[threadIdx.x] = ss;
  __syncthreads();
  for (int st = 128; st > 0; st >>= 1) {
    if (threadIdx.x < st) { shS[threadIdx.x] += shS[threadIdx.x + st]; shQ[threadIdx.x] += shQ[threadIdx.x + st]; }
    __syncthreads();
  }
  if (threadIdx.x == 0) {
    atomicAdd(&gnd[b], shS[0]);
    atomicAdd(&gnd[kB + b], shQ[0]);
  }
}

__global__ void gn_finalize_kernel(const double* __restrict__ s, const double* __restrict__ ss,
                                   float* __restrict__ mu, float* __restrict__ rsig) {
  int b = threadIdx.x;
  if (b < kB) {
    double N = (double)kL * kCH;
    double m = s[b] / N;
    double v = ss[b] / N - m * m;
    mu[b]   = (float)m;
    rsig[b] = (float)(1.0 / sqrt(v + (double)kEPS));
  }
}

// ---------------------------------------------------------------------------
// K3: qkv = gn1(x) @ W_qkv.  Tile 32 L-rows, 8 waves each own 3 N-tiles.
// q -> row-major f16 [B,L,128]; k,v -> transposed f16 [B,H,D,L] (contiguous
// per-lane v8h stores along L thanks to C-tile layout); atomic max of k*scale.
// ---------------------------------------------------------------------------
__global__ __launch_bounds__(256) void qkv_kernel(
    const float* __restrict__ x,
    const float* __restrict__ g1w, const float* __restrict__ g1b,
    const float* __restrict__ mu1, const float* __restrict__ rs1,
    const _Float16* __restrict__ wqkvp,
    _Float16* __restrict__ qbuf, _Float16* __restrict__ kT,
    _Float16* __restrict__ vT, unsigned* __restrict__ kmaxk) {
  __shared__ __align__(32) _Float16 xs[32 * kCH];
  int b  = blockIdx.y;
  int l0 = blockIdx.x * 32;
  int tid = threadIdx.x;
  float mu = mu1[b], rs = rs1[b];
  const float* xb = x + ((size_t)b * kL + l0) * kCH;
  for (int i = tid; i < 32 * kCH; i += 256) {
    int c = i & 127;
    xs[i] = (_Float16)((xb[i] - mu) * rs * g1w[c] + g1b[c]);
  }
  __syncthreads();

  int wave = tid >> 5, lane = tid & 31;
  v8f acc[2][3] = {};
#pragma unroll
  for (int kc = 0; kc < 4; ++kc) {
    v16h a0 = load_a_f16(xs, kCH, kc * 32, lane);
    v16h a1 = load_a_f16(xs + 16 * kCH, kCH, kc * 32, lane);
#pragma unroll
    for (int t = 0; t < 3; ++t) {
      int nt = wave * 3 + t;
      v16h bf = *(const v16h*)(wqkvp + (size_t)(kc * 24 + nt) * 512 + lane * 16);
      acc[0][t] = wmma_f16(a0, bf, acc[0][t]);
      acc[1][t] = wmma_f16(a1, bf, acc[1][t]);
    }
  }

  int n = lane & 15, mbase = (lane < 16) ? 0 : 8;
#pragma unroll
  for (int t = 0; t < 3; ++t) {
    int nt  = wave * 3 + t;
    int col = nt * 16 + n;
#pragma unroll
    for (int mt = 0; mt < 2; ++mt) {
      v8f c = acc[mt][t];
      int lbase = l0 + mt * 16 + mbase;
      if (col < kCH) {                       // q: row-major
        _Float16* qp = qbuf + ((size_t)b * kL + lbase) * kCH + col;
#pragma unroll
        for (int i = 0; i < 8; ++i) qp[(size_t)i * kCH] = (_Float16)c[i];
      } else if (col < 2 * kCH) {            // k: transposed + scaled + max
        int ck = col - kCH, h = ck >> 5, d = ck & 31;
        float mx = -3.0e38f;
        v8h pk;
#pragma unroll
        for (int i = 0; i < 8; ++i) {
          float s = c[i] * kKSCALE;
          pk[i] = (_Float16)s;
          mx = fmaxf(mx, s);
        }
        *(v8h*)(kT + ((size_t)((b * kH + h) * kD + d)) * kL + lbase) = pk;
        atomicMax(kmaxk + (b * kH + h) * kD + d, fkey(mx));
      } else {                               // v: transposed
        int cv = col - 2 * kCH, h = cv >> 5, d = cv & 31;
        v8h pv;
#pragma unroll
        for (int i = 0; i < 8; ++i) pv[i] = (_Float16)c[i];
        *(v8h*)(vT + ((size_t)((b * kH + h) * kD + d)) * kL + lbase) = pv;
      }
    }
  }
}

// ---------------------------------------------------------------------------
// K4: context[b,h] = softmax_num(k)^T @ v over L (D x D), WMMA over K=L.
// One wave per block; p=exp(k-max) built directly into A fragments.
// ---------------------------------------------------------------------------
__device__ inline v16h exp_a(const _Float16* p, int koff, float km, float& s) {
  v8h lo = *(const v8h*)(p + koff);
  v8h hi = *(const v8h*)(p + 8 + koff);
  v16h a;
#pragma unroll
  for (int i = 0; i < 8; ++i) {
    float e0 = __expf((float)lo[i] - km);
    float e1 = __expf((float)hi[i] - km);
    s += e0 + e1;
    a[i] = (_Float16)e0; a[8 + i] = (_Float16)e1;
  }
  return a;
}

constexpr int kNCHUNK = 64;

__global__ __launch_bounds__(32) void ctx_kernel(
    const _Float16* __restrict__ kT, const _Float16* __restrict__ vT,
    const unsigned* __restrict__ kmaxk,
    float* __restrict__ sumP, float* __restrict__ ctx) {
  int bh   = blockIdx.x;           // b*H + h
  int chnk = blockIdx.y;           // L chunk
  int lane = threadIdx.x;
  const int LC = kL / kNCHUNK;

  int m    = lane & 15;
  int koff = (lane < 16) ? 0 : 16; // A-fragment K offset
  int nn   = lane & 15;
  int bkoff = (lane < 16) ? 0 : 16; // B-fragment K offset
  float km0 = funkey(kmaxk[bh * kD + m]);
  float km1 = funkey(kmaxk[bh * kD + 16 + m]);

  const _Float16* k0p = kT + (size_t)(bh * kD + m) * kL;
  const _Float16* k1p = kT + (size_t)(bh * kD + 16 + m) * kL;
  const _Float16* vb  = vT + (size_t)bh * kD * kL;

  v8f c00 = {}, c01 = {}, c10 = {}, c11 = {};
  float sp0 = 0.f, sp1 = 0.f;
  for (int l0 = chnk * LC; l0 < (chnk + 1) * LC; l0 += 32) {
    v16h a0 = exp_a(k0p + l0, koff, km0, sp0);
    v16h a1 = exp_a(k1p + l0, koff, km1, sp1);
    v16h b0 = *(const v16h*)(vb + (size_t)nn * kL + l0 + bkoff);
    v16h b1 = *(const v16h*)(vb + (size_t)(16 + nn) * kL + l0 + bkoff);
    c00 = wmma_f16(a0, b0, c00);
    c01 = wmma_f16(a0, b1, c01);
    c10 = wmma_f16(a1, b0, c10);
    c11 = wmma_f16(a1, b1, c11);
  }
  // combine K-half partial sums (lanes t and t+16 cover complementary K)
  sp0 += __shfl_xor(sp0, 16, 32);
  sp1 += __shfl_xor(sp1, 16, 32);
  if (lane < 16) {
    atomicAdd(&sumP[bh * kD + lane], sp0);
    atomicAdd(&sumP[bh * kD + 16 + lane], sp1);
  }
  int n = lane & 15, mb = (lane < 16) ? 0 : 8;
  float* cp = ctx + (size_t)bh * kD * kD;
#pragma unroll
  for (int i = 0; i < 8; ++i) {
    atomicAdd(&cp[(mb + i) * kD + n],            c00[i]);
    atomicAdd(&cp[(mb + i) * kD + 16 + n],       c01[i]);
    atomicAdd(&cp[(16 + mb + i) * kD + n],       c10[i]);
    atomicAdd(&cp[(16 + mb + i) * kD + 16 + n],  c11[i]);
  }
}

// ---------------------------------------------------------------------------
// K5: normalize context rows by sumP and pre-pack into B-fragment f16.
// ---------------------------------------------------------------------------
__global__ __launch_bounds__(256) void ctxpack_kernel(
    const float* __restrict__ ctx, const float* __restrict__ sumP,
    _Float16* __restrict__ ctxp) {
  int id = blockIdx.x * 256 + threadIdx.x;       // 32768 total
  int bh = id >> 10, t = id & 1023;
  int nt = t >> 9, r = t & 511, lane = r >> 4, j = r & 15;
  int k = ((lane < 16) ? 0 : 16) + j;
  int n = nt * 16 + (lane & 15);
  ctxp[id] = (_Float16)(ctx[bh * kD * kD + k * kD + n] / sumP[bh * kD + k]);
}

// ---------------------------------------------------------------------------
// K6/K7: att = softmax_D(q) @ ctx ; y = att @ W_out.
// STATS pass accumulates GN2 mean/var; FINAL pass recomputes y and writes
// gn2(y) + x  (recompute is cheaper than a 268MB f32 y round-trip).
// ---------------------------------------------------------------------------
template <bool FINAL>
__global__ __launch_bounds__(128) void attout_kernel(
    const _Float16* __restrict__ qbuf, const _Float16* __restrict__ ctxp,
    const _Float16* __restrict__ woutp,
    const float* __restrict__ x,
    const float* __restrict__ g2w, const float* __restrict__ g2b,
    const float* __restrict__ mu2, const float* __restrict__ rs2,
    double* __restrict__ gnd2, float* __restrict__ out) {
  __shared__ __align__(32) _Float16 qs[32 * kCH];
  __shared__ __align__(32) _Float16 atts[32 * kCH];
  __shared__ float redS[128], redQ[128];

  int b   = blockIdx.y;
  int l0  = blockIdx.x * 32;
  int tid = threadIdx.x;
  int wave = tid >> 5, lane = tid & 31;

  // phase 1: q softmax over D per (l, h); thread = one (l, h) pair
  {
    int l = tid & 31, h = tid >> 5;
    const _Float16* qp = qbuf + ((size_t)b * kL + l0 + l) * kCH + h * kD;
    float v[kD];
    float mx = -3.0e38f;
#pragma unroll
    for (int d = 0; d < kD; ++d) { v[d] = (float)qp[d] * kQSCALE; mx = fmaxf(mx, v[d]); }
    float s = 0.f;
#pragma unroll
    for (int d = 0; d < kD; ++d) { v[d] = __expf(v[d] - mx); s += v[d]; }
    float inv = 1.f / s;
#pragma unroll
    for (int d = 0; d < kD; ++d) qs[l * kCH + h * kD + d] = (_Float16)(v[d] * inv);
  }
  __syncthreads();

  // phase 2: att[:, h*32:(h+1)*32] = qs_h(32x32) @ ctx_h(32x32); wave = head
  {
    int h = wave;
    v16h a0 = load_a_f16(qs, kCH, h * kD, lane);
    v16h a1 = load_a_f16(qs + 16 * kCH, kCH, h * kD, lane);
    const _Float16* cp = ctxp + (size_t)(b * kH + h) * 1024;
    v16h b0 = *(const v16h*)(cp + lane * 16);
    v16h b1 = *(const v16h*)(cp + 512 + lane * 16);
    v8f d00 = {}, d01 = {}, d10 = {}, d11 = {};
    d00 = wmma_f16(a0, b0, d00);
    d01 = wmma_f16(a0, b1, d01);
    d10 = wmma_f16(a1, b0, d10);
    d11 = wmma_f16(a1, b1, d11);
    int n = lane & 15, mb = (lane < 16) ? 0 : 8;
#pragma unroll
    for (int i = 0; i < 8; ++i) {
      atts[(mb + i) * kCH + h * kD + n]           = (_Float16)d00[i];
      atts[(mb + i) * kCH + h * kD + 16 + n]      = (_Float16)d01[i];
      atts[(16 + mb + i) * kCH + h * kD + n]      = (_Float16)d10[i];
      atts[(16 + mb + i) * kCH + h * kD + 16 + n] = (_Float16)d11[i];
    }
  }
  __syncthreads();

  // phase 3: y = att(32x128) @ W_out(128x128); wave owns N-tiles {w, w+4}
  v8f y[2][2] = {};
  int nts[2] = {wave, wave + 4};
#pragma unroll
  for (int kc = 0; kc < 4; ++kc) {
    v16h a0 = load_a_f16(atts, kCH, kc * 32, lane);
    v16h a1 = load_a_f16(atts + 16 * kCH, kCH, kc * 32, lane);
#pragma unroll
    for (int j = 0; j < 2; ++j) {
      v16h bf = *(const v16h*)(woutp + (size_t)(kc * 8 + nts[j]) * 512 + lane * 16);
      y[0][j] = wmma_f16(a0, bf, y[0][j]);
      y[1][j] = wmma_f16(a1, bf, y[1][j]);
    }
  }

  int n = lane & 15, mb = (lane < 16) ? 0 : 8;
  if (!FINAL) {
    float s = 0.f, ss = 0.f;
#pragma unroll
    for (int mt = 0; mt < 2; ++mt)
#pragma unroll
      for (int j = 0; j < 2; ++j)
#pragma unroll
        for (int i = 0; i < 8; ++i) { float v = y[mt][j][i]; s += v; ss += v * v; }
    redS[tid] = s; redQ[tid] = ss;
    __syncthreads();
    for (int st = 64; st > 0; st >>= 1) {
      if (tid < st) { redS[tid] += redS[tid + st]; redQ[tid] += redQ[tid + st]; }
      __syncthreads();
    }
    if (tid == 0) {
      atomicAdd(&gnd2[b], (double)redS[0]);
      atomicAdd(&gnd2[kB + b], (double)redQ[0]);
    }
  } else {
    float m2 = mu2[b], r2 = rs2[b];
#pragma unroll
    for (int mt = 0; mt < 2; ++mt)
#pragma unroll
      for (int j = 0; j < 2; ++j) {
        int ch = nts[j] * 16 + n;
        float w = g2w[ch], bb = g2b[ch];
#pragma unroll
        for (int i = 0; i < 8; ++i) {
          int l = l0 + mt * 16 + mb + i;
          size_t idx = ((size_t)b * kL + l) * kCH + ch;
          out[idx] = (y[mt][j][i] - m2) * r2 * w + bb + x[idx];
        }
      }
  }
}

// ---------------------------------------------------------------------------
extern "C" void kernel_launch(void* const* d_in, const int* in_sizes, int n_in,
                              void* d_out, int out_size, void* d_ws, size_t ws_size,
                              hipStream_t stream) {
  const float* x    = (const float*)d_in[0];
  const float* g1w  = (const float*)d_in[1];
  const float* g1b  = (const float*)d_in[2];
  const float* Wqkv = (const float*)d_in[3];
  const float* Wout = (const float*)d_in[4];
  const float* g2w  = (const float*)d_in[5];
  const float* g2b  = (const float*)d_in[6];
  float* out = (float*)d_out;

  // workspace carve-up (256B aligned)
  char* ws = (char*)d_ws;
  auto alloc = [&](size_t bytes) -> void* {
    void* p = (void*)ws;
    ws += (bytes + 255) & ~(size_t)255;
    return p;
  };
  _Float16* wqkvp = (_Float16*)alloc((size_t)4 * 24 * 512 * 2);          // 96KB
  _Float16* woutp = (_Float16*)alloc((size_t)4 * 8 * 512 * 2);           // 32KB
  _Float16* kT    = (_Float16*)alloc((size_t)kB * kH * kD * kL * 2);     // 64MB
  _Float16* vT    = (_Float16*)alloc((size_t)kB * kH * kD * kL * 2);     // 64MB
  _Float16* qbuf  = (_Float16*)alloc((size_t)kB * kL * kCH * 2);         // 64MB
  float*    ctx   = (float*)alloc((size_t)kB * kH * kD * kD * 4);        // 128KB
  _Float16* ctxp  = (_Float16*)alloc((size_t)kB * kH * kD * kD * 2);     // 64KB
  float*    sumP  = (float*)alloc((size_t)kB * kH * kD * 4);
  unsigned* kmaxk = (unsigned*)alloc((size_t)kB * kH * kD * 4);
  double*   gnd   = (double*)alloc((size_t)4 * kB * 8);   // [sum1|ss1|sum2|ss2]
  float*    mus   = (float*)alloc((size_t)4 * kB * 4);    // [mu1|rs1|mu2|rs2]

  // 0) init + weight pack
  init_kernel<<<192, 256, 0, stream>>>(Wqkv, Wout, wqkvp, woutp, ctx, sumP, kmaxk, gnd);
  // 1) GN1 stats
  gn_stats_kernel<<<dim3(64, kB), 256, 0, stream>>>(x, gnd);
  gn_finalize_kernel<<<1, 32, 0, stream>>>(gnd, gnd + kB, mus, mus + kB);
  // 2) qkv GEMM (WMMA), transposed k/v stores, k max
  qkv_kernel<<<dim3(kL / 32, kB), 256, 0, stream>>>(
      x, g1w, g1b, mus, mus + kB, wqkvp, qbuf, kT, vT, kmaxk);
  // 3) context = softmax_num(k)^T @ v (WMMA over L)
  ctx_kernel<<<dim3(kB * kH, kNCHUNK), 32, 0, stream>>>(kT, vT, kmaxk, sumP, ctx);
  // 4) normalize + pack context
  ctxpack_kernel<<<kB * kH * kD * kD / 256, 256, 0, stream>>>(ctx, sumP, ctxp);
  // 5) att + W_out GEMMs: stats pass, finalize GN2, final pass + residual
  attout_kernel<false><<<dim3(kL / 32, kB), 128, 0, stream>>>(
      qbuf, ctxp, woutp, x, g2w, g2b, mus + 2 * kB, mus + 3 * kB, gnd + 2 * kB, out);
  gn_finalize_kernel<<<1, 32, 0, stream>>>(gnd + 2 * kB, gnd + 3 * kB, mus + 2 * kB, mus + 3 * kB);
  attout_kernel<true><<<dim3(kL / 32, kB), 128, 0, stream>>>(
      qbuf, ctxp, woutp, x, g2w, g2b, mus + 2 * kB, mus + 3 * kB, gnd + 2 * kB, out);
}
